// MultiHeadSelfAttention_24541443130010
// MI455X (gfx1250) — matrix operand order repaired and checked by reading, not compiled, not run on hardware
//
#include <hip/hip_runtime.h>
#include <hip/hip_bf16.h>

// ---------------------------------------------------------------------------
// MI455X (gfx1250) GQA attention block, bf16 WMMA pipeline.
//   B=4, S=2048, D=2048, H_Q=16, H_KV=4, D_HEAD=128, causal mask.
// ~310 GFLOP total; matrix-op bound on MI455X -> all GEMM-shaped work goes
// through v_wmma_f32_16x16x32_bf16 (f32 accumulate). GEMMs are double-buffered
// with CDNA5 async global->LDS copies (ASYNCcnt) to keep the matrix pipe fed.
// ---------------------------------------------------------------------------

#define SSEQ   2048
#define DMODEL 2048
#define NQKV   3072
#define DH     128
#define NHQ    16
#define NHKV   4
#define BATCH  4

typedef __attribute__((ext_vector_type(16))) __bf16 v16bf;
typedef __attribute__((ext_vector_type(8)))  __bf16 v8bf;
typedef __attribute__((ext_vector_type(8)))  float  v8f;
typedef __attribute__((ext_vector_type(4)))  float  v4f;
typedef __attribute__((ext_vector_type(4)))  int    v4i;

union V16u { v16bf v; v8bf h[2]; };

#define WMMA_BF16(Afr, Bfr, Cfr) \
  __builtin_amdgcn_wmma_f32_16x16x32_bf16(false, (Afr), false, (Bfr), (short)0, (Cfr), false, false)

// ---------------------------------------------------------------------------
// CDNA5 async global->LDS copy (16B per lane), tracked by ASYNCcnt.
// Builtin signature (probe-confirmed by round-2 diagnostic):
//   (v4i addrspace(1)*, v4i addrspace(3)*, imm int offset, imm int cpol)
// ---------------------------------------------------------------------------
__device__ inline void async_copy16(const __bf16* g, __bf16* l) {
#if defined(__AMDGCN__)
# if __has_builtin(__builtin_amdgcn_global_load_async_to_lds_b128)
  __builtin_amdgcn_global_load_async_to_lds_b128(
      (__attribute__((address_space(1))) v4i*)(unsigned long long)(size_t)g,
      (__attribute__((address_space(3))) v4i*)(unsigned)(size_t)l, 0, 0);
# else
  // VDST = LDS byte offset (low 32 bits of the generic LDS pointer),
  // VADDR = 64-bit global address.
  asm volatile("global_load_async_to_lds_b128 %0, %1, off"
               :
               : "v"((unsigned)(size_t)l), "v"((unsigned long long)(size_t)g)
               : "memory");
# endif
#else
  (void)g; (void)l;
#endif
}

__device__ inline void wait_async0() {
#if defined(__AMDGCN__)
# if __has_builtin(__builtin_amdgcn_s_wait_asynccnt)
  __builtin_amdgcn_s_wait_asynccnt(0);
# else
  asm volatile("s_wait_asynccnt 0" ::: "memory");
# endif
#endif
}

// A-fragment (16 rows x 32 K, bf16) from an LDS tile, row-major, stride elems.
// Lane layout (ISA 7.12.2): lanes 0-15 row m hold K 0-7 & 16-23; lanes 16-31
// hold K 8-15 & 24-31.
__device__ inline v16bf load_a16x32(const __bf16* tile, int stride) {
  int lane = threadIdx.x & 31;
  int m = lane & 15, hh = lane >> 4;
  const __bf16* r = tile + m * stride;
  V16u u;
  u.h[0] = *(const v8bf*)(r + hh * 8);
  u.h[1] = *(const v8bf*)(r + 16 + hh * 8);
  return u.v;
}

// B-fragment (32 K x 16 N, bf16) from a TRANSPOSED LDS tile (tileT[n][k]).
// Lane n holds column n; lanes 0-15 K 0-7 & 16-23; lanes 16-31 K 8-15 & 24-31.
__device__ inline v16bf load_b32x16(const __bf16* tileT, int stride) {
  int lane = threadIdx.x & 31;
  int n = lane & 15, hh = lane >> 4;
  const __bf16* r = tileT + n * stride;
  V16u u;
  u.h[0] = *(const v8bf*)(r + hh * 8);
  u.h[1] = *(const v8bf*)(r + 16 + hh * 8);
  return u.v;
}

// ---------------------------------------------------------------------------
// Kernel 1: f32 -> bf16 convert (x)
// ---------------------------------------------------------------------------
__global__ __launch_bounds__(256) void f32_to_bf16_kernel(
    const float* __restrict__ in, __bf16* __restrict__ out, size_t n) {
  size_t i = ((size_t)blockIdx.x * 256 + threadIdx.x) * 8;
  if (i + 8 <= n) {
    v4f a = *(const v4f*)(in + i);
    v4f b = *(const v4f*)(in + i + 4);
#pragma unroll
    for (int j = 0; j < 4; ++j) out[i + j]     = (__bf16)a[j];
#pragma unroll
    for (int j = 0; j < 4; ++j) out[i + 4 + j] = (__bf16)b[j];
  }
}

// ---------------------------------------------------------------------------
// Kernel 2/5: GEMM  C[M,N] = A(bf16,MxK) @ B(f32->bf16,KxN)
// 128x128 block tile, 256 threads = 8 waves in 4(M) x 2(N), wave tile 32x64,
// K chunk 32, 8 WMMA per wave per chunk. Double-buffered LDS: A tile staged
// via async global->LDS DMA, B tile prefetched through VGPRs (needs f32->bf16
// convert + transpose).
// ---------------------------------------------------------------------------

// Issue async staging of a 128x32 bf16 A tile into dstAs (stride 40 elems).
__device__ inline void stage_a_async(const __bf16* __restrict__ A, int lda,
                                     int bm, int k0, __bf16* dstAs) {
  int tid = threadIdx.x;
  int r = tid >> 1, c0 = (tid & 1) * 16;               // 128 rows x 32 cols
  const __bf16* g = A + (size_t)(bm + r) * lda + k0 + c0;
  __bf16* l = dstAs + r * 40 + c0;
  async_copy16(g, l);
  async_copy16(g + 8, l + 8);
}

__device__ inline void load_b_regs(const float* __restrict__ B, int ldb,
                                   int bn, int k0, v4f regs[4]) {
  int tid = threadIdx.x;
  int kr = tid >> 3, c0 = (tid & 7) * 16;              // 32 rows x 128 cols
  const float* src = B + (size_t)(k0 + kr) * ldb + bn + c0;
#pragma unroll
  for (int i = 0; i < 4; ++i) regs[i] = *(const v4f*)(src + 4 * i);
}

__device__ inline void store_b_lds(__bf16* dstBt, const v4f regs[4]) {
  int tid = threadIdx.x;
  int kr = tid >> 3, c0 = (tid & 7) * 16;
#pragma unroll
  for (int i = 0; i < 4; ++i)
#pragma unroll
    for (int j = 0; j < 4; ++j)
      dstBt[(c0 + 4 * i + j) * 40 + kr] = (__bf16)regs[i][j];
}

template <typename TOut>
__global__ __launch_bounds__(256) void gemm_bf16_kernel(
    const __bf16* __restrict__ A, const float* __restrict__ B,
    TOut* __restrict__ C, int M, int N, int K) {
  __shared__ __attribute__((aligned(16))) __bf16 As[2][128 * 40];   // [m][k]
  __shared__ __attribute__((aligned(16))) __bf16 Bt[2][128 * 40];   // [n][k]

  int tid = threadIdx.x;
  int lane = tid & 31, wid = tid >> 5;
  int wm = wid & 3, wn = wid >> 2;             // 4 waves M, 2 waves N
  int bm = blockIdx.x * 128, bn = blockIdx.y * 128;

  v8f acc[2][4];
#pragma unroll
  for (int i = 0; i < 2; ++i)
#pragma unroll
    for (int j = 0; j < 4; ++j) acc[i][j] = (v8f){0, 0, 0, 0, 0, 0, 0, 0};

  const int nk = K >> 5;

  // Prologue: stage chunk 0 into buffer 0.
  stage_a_async(A, K, bm, 0, &As[0][0]);
  {
    v4f b0[4];
    load_b_regs(B, N, bn, 0, b0);
    store_b_lds(&Bt[0][0], b0);
  }
  wait_async0();
  __syncthreads();

  for (int kc = 0; kc < nk; ++kc) {
    int buf = kc & 1;
    bool hasnext = (kc + 1) < nk;

    // Prefetch next chunk: async A -> LDS[buf^1], B -> VGPRs.
    v4f bregs[4];
    if (hasnext) {
      stage_a_async(A, K, bm, (kc + 1) * 32, &As[buf ^ 1][0]);
      load_b_regs(B, N, bn, (kc + 1) * 32, bregs);
    }

    // Compute on current buffer: 8 v_wmma_f32_16x16x32_bf16 per wave.
    v16bf afr[2], bfr[4];
#pragma unroll
    for (int mt = 0; mt < 2; ++mt)
      afr[mt] = load_a16x32(&As[buf][0] + (wm * 32 + mt * 16) * 40, 40);
#pragma unroll
    for (int nt = 0; nt < 4; ++nt)
      bfr[nt] = load_b32x16(&Bt[buf][0] + (wn * 64 + nt * 16) * 40, 40);
#pragma unroll
    for (int mt = 0; mt < 2; ++mt)
#pragma unroll
      for (int nt = 0; nt < 4; ++nt)
        acc[mt][nt] = WMMA_BF16(afr[mt], bfr[nt], acc[mt][nt]);

    if (hasnext) {
      store_b_lds(&Bt[buf ^ 1][0], bregs);
      wait_async0();
    }
    __syncthreads();
  }

  // Epilogue: C layout slot r -> row r + 8*half, lane -> column.
  int n = lane & 15, hh = lane >> 4;
#pragma unroll
  for (int mt = 0; mt < 2; ++mt)
#pragma unroll
    for (int nt = 0; nt < 4; ++nt)
#pragma unroll
      for (int r = 0; r < 8; ++r) {
        int row = bm + wm * 32 + mt * 16 + r + 8 * hh;
        int col = bn + wn * 64 + nt * 16 + n;
        C[(size_t)row * N + col] = static_cast<TOut>(acc[mt][nt][r]);
      }
}

// ---------------------------------------------------------------------------
// Kernel 3: RoPE + split qkv[tok][3072] -> Q/K/V in [B,H,S,128] bf16.
// Folds the 1/sqrt(128) softmax scale into Q.
// ---------------------------------------------------------------------------
__global__ __launch_bounds__(256) void rope_split_kernel(
    const __bf16* __restrict__ qkv, __bf16* __restrict__ Qo,
    __bf16* __restrict__ Ko, __bf16* __restrict__ Vo) {
  int tok = blockIdx.x;
  int b = tok / SSEQ, s = tok % SSEQ;
  const __bf16* row = qkv + (size_t)tok * NQKV;
  const float scale = 0.08838834764831845f;        // 1/sqrt(128)
  const float l2_10000 = 13.287712379549449f;      // log2(10000)

  for (int p = threadIdx.x; p < NQKV / 2; p += 256) {
    int c = 2 * p;
    if (c < NHQ * DH) {                            // Q heads
      int hq = c / DH, d = c % DH;
      float theta = exp2f(-(float)d * (l2_10000 / (float)DH));
      float ang = (float)s * theta;
      float cs = cosf(ang), sn = sinf(ang);
      float x0 = (float)row[c], x1 = (float)row[c + 1];
      size_t o = ((size_t)(b * NHQ + hq) * SSEQ + s) * DH + d;
      Qo[o]     = (__bf16)((x0 * cs - x1 * sn) * scale);
      Qo[o + 1] = (__bf16)((x1 * cs + x0 * sn) * scale);
    } else if (c < (NHQ + NHKV) * DH) {            // K heads
      int cc = c - NHQ * DH;
      int hk = cc / DH, d = cc % DH;
      float theta = exp2f(-(float)d * (l2_10000 / (float)DH));
      float ang = (float)s * theta;
      float cs = cosf(ang), sn = sinf(ang);
      float x0 = (float)row[c], x1 = (float)row[c + 1];
      size_t o = ((size_t)(b * NHKV + hk) * SSEQ + s) * DH + d;
      Ko[o]     = (__bf16)(x0 * cs - x1 * sn);
      Ko[o + 1] = (__bf16)(x1 * cs + x0 * sn);
    } else {                                       // V heads (copy)
      int cc = c - (NHQ + NHKV) * DH;
      int hv = cc / DH, d = cc % DH;
      size_t o = ((size_t)(b * NHKV + hv) * SSEQ + s) * DH + d;
      Vo[o]     = row[c];
      Vo[o + 1] = row[c + 1];
    }
  }
}

// ---------------------------------------------------------------------------
// Kernel 4: causal flash attention, bf16 WMMA, online softmax.
// Grid (S/64, H_Q, B), 128 threads = 4 waves; wave w owns q rows
// [q0+16w, q0+16w+16). KV processed in 32-wide blocks; V staged transposed
// in LDS (shared), K fragments loaded straight from global (K^T columns are
// contiguous K rows).
// ---------------------------------------------------------------------------
__global__ __launch_bounds__(128) void flash_attn_kernel(
    const __bf16* __restrict__ Q, const __bf16* __restrict__ Kg,
    const __bf16* __restrict__ Vg, __bf16* __restrict__ O) {
  __shared__ __attribute__((aligned(16))) __bf16 VT[128 * 40];       // [d][kv]
  __shared__ __attribute__((aligned(16))) __bf16 Pb[4][16 * 40];     // per wave

  int tid = threadIdx.x, w = tid >> 5, lane = tid & 31;
  int n = lane & 15, hh = lane >> 4;
  int q0 = blockIdx.x * 64;
  int h = blockIdx.y, b = blockIdx.z, hkv = h >> 2;                  // G = 4

  const __bf16* qbase = Q + ((size_t)(b * NHQ + h) * SSEQ) * DH;
  const __bf16* kbase = Kg + ((size_t)(b * NHKV + hkv) * SSEQ) * DH;
  const __bf16* vbase = Vg + ((size_t)(b * NHKV + hkv) * SSEQ) * DH;

  // Q fragments for this wave's 16 rows (A-frag lane layout), kept resident.
  int qrow = q0 + w * 16 + n;
  v16bf qf[4];
#pragma unroll
  for (int dc = 0; dc < 4; ++dc) {
    const __bf16* r = qbase + (size_t)qrow * DH + dc * 32;
    V16u u;
    u.h[0] = *(const v8bf*)(r + hh * 8);
    u.h[1] = *(const v8bf*)(r + 16 + hh * 8);
    qf[dc] = u.v;
  }

  v8f o_acc[8];
#pragma unroll
  for (int t = 0; t < 8; ++t) o_acc[t] = (v8f){0, 0, 0, 0, 0, 0, 0, 0};
  float mrow[8], lrow[8];
#pragma unroll
  for (int r = 0; r < 8; ++r) { mrow[r] = -1e30f; lrow[r] = 0.f; }

  int nblk = (q0 + 64) >> 5;                       // causal KV extent
  for (int jb = 0; jb < nblk; ++jb) {
    int j = jb * 32;

    // Cooperative V^T staging: 32 kv x 128 d -> VT[d][kv].
    {
      int kv = tid >> 2, d0 = (tid & 3) * 32;
      const v8bf* src = (const v8bf*)(vbase + (size_t)(j + kv) * DH + d0);
#pragma unroll
      for (int q4 = 0; q4 < 4; ++q4) {
        v8bf chunk = src[q4];
#pragma unroll
        for (int e = 0; e < 8; ++e) VT[(d0 + q4 * 8 + e) * 40 + kv] = chunk[e];
      }
    }
    __syncthreads();

    // Scores: two 16-col tiles, 4 WMMA each over the 128-dim contraction.
    v8f s[2];
    s[0] = (v8f){0, 0, 0, 0, 0, 0, 0, 0};
    s[1] = (v8f){0, 0, 0, 0, 0, 0, 0, 0};
#pragma unroll
    for (int nt = 0; nt < 2; ++nt) {
#pragma unroll
      for (int dc = 0; dc < 4; ++dc) {
        // B-frag for Q@K^T: lane n holds K row (kv = j+nt*16+n), contiguous.
        const __bf16* r = kbase + (size_t)(j + nt * 16 + n) * DH + dc * 32;
        V16u u;
        u.h[0] = *(const v8bf*)(r + hh * 8);
        u.h[1] = *(const v8bf*)(r + 16 + hh * 8);
        s[nt] = WMMA_BF16(qf[dc], u.v, s[nt]);
      }
    }

    // Causal mask (only diagonal-crossing blocks hit this path).
    if (j + 31 > q0 + w * 16) {
#pragma unroll
      for (int nt = 0; nt < 2; ++nt)
#pragma unroll
        for (int r = 0; r < 8; ++r) {
          int row = q0 + w * 16 + r + 8 * hh;
          int kv = j + nt * 16 + n;
          if (kv > row) s[nt][r] = -1e30f;
        }
    }

    // Online softmax per slot row; reductions across the 16-lane half.
#pragma unroll
    for (int r = 0; r < 8; ++r) {
      float mx = fmaxf(s[0][r], s[1][r]);
#pragma unroll
      for (int off = 8; off > 0; off >>= 1) mx = fmaxf(mx, __shfl_xor(mx, off, 32));
      float mnew = fmaxf(mrow[r], mx);
      float alpha = expf(mrow[r] - mnew);
      float p0 = expf(s[0][r] - mnew);
      float p1 = expf(s[1][r] - mnew);
      float ps = p0 + p1;
#pragma unroll
      for (int off = 8; off > 0; off >>= 1) ps += __shfl_xor(ps, off, 32);
      lrow[r] = lrow[r] * alpha + ps;
      mrow[r] = mnew;
#pragma unroll
      for (int t = 0; t < 8; ++t) o_acc[t][r] *= alpha;
      // C-layout -> A-frag layout rearrange via wave-private LDS strip.
      Pb[w][(r + 8 * hh) * 40 + n]      = (__bf16)p0;
      Pb[w][(r + 8 * hh) * 40 + 16 + n] = (__bf16)p1;
    }

    // P (16x32) @ V (32x128): 8 WMMA across head-dim tiles.
    v16bf pf = load_a16x32(&Pb[w][0], 40);
#pragma unroll
    for (int t = 0; t < 8; ++t) {
      v16bf vf = load_b32x16(VT + (t * 16) * 40, 40);
      o_acc[t] = WMMA_BF16(pf, vf, o_acc[t]);
    }
    __syncthreads();   // protect VT before next block's staging
  }

  // Epilogue: normalize, write [b, s, h*128 + d] bf16 (A-matrix for Wo GEMM).
#pragma unroll
  for (int r = 0; r < 8; ++r) {
    float inv = 1.0f / lrow[r];
    int row = q0 + w * 16 + r + 8 * hh;
    __bf16* dst = O + ((size_t)(b * SSEQ + row)) * DMODEL + h * DH;
#pragma unroll
    for (int t = 0; t < 8; ++t) dst[t * 16 + n] = (__bf16)(o_acc[t][r] * inv);
  }
}

// ---------------------------------------------------------------------------
// Launch. Workspace layout (bytes, total ~168MB):
//   [0)           x_bf16     8192*2048*2 = 33,554,432
//   [33554432)    qkv_bf16   8192*3072*2 = 50,331,648
//   [83886080)    q_bf16     4*16*2048*128*2 = 33,554,432
//   [117440512)   k_bf16     4*4*2048*128*2  =  8,388,608
//   [125829120)   v_bf16                     =  8,388,608
//   [134217728)   attn_out   8192*2048*2     = 33,554,432
// ---------------------------------------------------------------------------
extern "C" void kernel_launch(void* const* d_in, const int* in_sizes, int n_in,
                              void* d_out, int out_size, void* d_ws, size_t ws_size,
                              hipStream_t stream) {
  const float* x    = (const float*)d_in[0];
  const float* Wqkv = (const float*)d_in[1];
  const float* Wo   = (const float*)d_in[2];
  // d_in[3] = causal mask: causality is hardcoded in flash_attn_kernel.
  float* out = (float*)d_out;

  char* ws = (char*)d_ws;
  __bf16* x_bf   = (__bf16*)(ws);
  __bf16* qkv_bf = (__bf16*)(ws + 33554432ull);
  __bf16* q_bf   = (__bf16*)(ws + 83886080ull);
  __bf16* k_bf   = (__bf16*)(ws + 117440512ull);
  __bf16* v_bf   = (__bf16*)(ws + 125829120ull);
  __bf16* ao_bf  = (__bf16*)(ws + 134217728ull);

  const size_t n_x = (size_t)BATCH * SSEQ * DMODEL;   // 16,777,216

  f32_to_bf16_kernel<<<(int)(n_x / (256 * 8)), 256, 0, stream>>>(x, x_bf, n_x);

  // qkv = x @ Wqkv : M=8192, N=3072, K=2048
  gemm_bf16_kernel<__bf16><<<dim3(8192 / 128, NQKV / 128), 256, 0, stream>>>(
      x_bf, Wqkv, qkv_bf, BATCH * SSEQ, NQKV, DMODEL);

  rope_split_kernel<<<BATCH * SSEQ, 256, 0, stream>>>(qkv_bf, q_bf, k_bf, v_bf);

  flash_attn_kernel<<<dim3(SSEQ / 64, NHQ, BATCH), 128, 0, stream>>>(
      q_bf, k_bf, v_bf, ao_bf);

  // out = attn_out @ Wo : M=8192, N=2048, K=2048
  gemm_bf16_kernel<float><<<dim3(8192 / 128, DMODEL / 128), 256, 0, stream>>>(
      ao_bf, Wo, out, BATCH * SSEQ, DMODEL, DMODEL);
}
